// net_a_bit_overkill_19997367730394
// MI455X (gfx1250) — matrix operand order repaired and checked by reading, not compiled, not run on hardware
//
#include <hip/hip_runtime.h>
#include <hip/hip_bf16.h>

typedef _Float16 h8   __attribute__((ext_vector_type(8)));
typedef _Float16 v16h __attribute__((ext_vector_type(16)));
typedef float    v8f  __attribute__((ext_vector_type(8)));

#define WMMA_F16(A, B, C) \
    __builtin_amdgcn_wmma_f32_16x16x32_f16(false, (A), false, (B), (short)0, (C), false, false)

__device__ __forceinline__ float relu_med3(float x) {
    // clamp to [0, +inf) == relu, single v_med3_num_f32
    return __builtin_amdgcn_fmed3f(x, 0.0f, __builtin_inff());
}

// ---------------------------------------------------------------------------
// Kernel 1: top-k magnitude threshold via binary search on |f| bit patterns.
// One block. Finds largest T such that count(magbits >= T) >= k  ==  bits of
// the k-th largest magnitude. Reference keeps |W| >= thr (ties included).
// ---------------------------------------------------------------------------
__global__ void topkast_threshold_kernel(const float* __restrict__ W, int n, int k,
                                         unsigned* __restrict__ thr_out) {
    __shared__ unsigned s_count;
    const int tid = threadIdx.x;
    unsigned lo = 0u, hi = 0x7fffffffu;
    while (lo < hi) {
        unsigned mid = lo + ((hi - lo + 1u) >> 1);
        if (tid == 0) s_count = 0u;
        __syncthreads();
        unsigned c = 0u;
        for (int i = tid; i < n; i += blockDim.x) {
            unsigned mag = __float_as_uint(W[i]) & 0x7fffffffu;
            c += (mag >= mid) ? 1u : 0u;
        }
        #pragma unroll
        for (int off = 1; off < 32; off <<= 1) c += __shfl_xor(c, off, 32);
        if ((tid & 31) == 0) atomicAdd(&s_count, c);
        __syncthreads();
        unsigned cnt = s_count;
        __syncthreads();
        if (cnt >= (unsigned)k) lo = mid; else hi = mid - 1u;
    }
    if (tid == 0) thr_out[0] = lo;
}

// ---------------------------------------------------------------------------
// Kernel 2: mask + pack W1/W2 into f16 WMMA B-fragments (per-lane 16 halves).
// B packing used here (contiguous K, matches A): lanes 0-15 -> K=0..15, lanes 16-31 -> K=16..31,
// N = lane%16; halves ordered by increasing K within the lane.
// W1B: [8 n-tiles][32 lanes][16 halves]   (K padded 13 -> 32 with zeros)
// W2B: [8 n-tiles][4 k-chunks][32 lanes][16 halves]
// ---------------------------------------------------------------------------
__global__ void topkast_pack_kernel(const float* __restrict__ W1, const float* __restrict__ W2,
                                    const unsigned* __restrict__ thr,
                                    _Float16* __restrict__ W1B, _Float16* __restrict__ W2B) {
    int idx = blockIdx.x * blockDim.x + threadIdx.x;
    if (idx < 4096) {                       // W1 fragments
        int t    = idx >> 9;
        int rem  = idx & 511;
        int lane = rem >> 4;
        int h16  = rem & 15;
        int n    = t * 16 + (lane & 15);
        int K    = ((lane < 16) ? 0 : 16) + h16;
        float v = 0.f;
        if (K < 13) {
            float w = W1[n * 13 + K];
            unsigned mag = __float_as_uint(w) & 0x7fffffffu;
            if (mag >= thr[0]) v = w;
        }
        W1B[idx] = (_Float16)v;
    } else if (idx < 4096 + 16384) {        // W2 fragments
        int i2   = idx - 4096;
        int t    = i2 >> 11;
        int rem  = i2 & 2047;
        int kc   = rem >> 9;
        int rem2 = rem & 511;
        int lane = rem2 >> 4;
        int h16  = rem2 & 15;
        int n    = t * 16 + (lane & 15);
        int K    = kc * 32 + ((lane < 16) ? 0 : 16) + h16;
        float w  = W2[n * 128 + K];
        unsigned mag = __float_as_uint(w) & 0x7fffffffu;
        W2B[i2] = (mag >= thr[1]) ? (_Float16)w : (_Float16)0.f;
    }
}

// ---------------------------------------------------------------------------
// Kernel 3: fused 3-layer MLP. 256 threads = 8 waves; 128 rows/block.
// Per wave (16 rows): 8 WMMAs (layer1, K padded to 32) + 32 WMMAs (layer2,
// K=128). Accumulators start as inline-0 SRC2; bias+relu fused in epilogue
// via v_med3. Layer-2 A-fragments hoisted out of the n-tile loop.
// ---------------------------------------------------------------------------
#define XS_PITCH 40    // halves per X row in LDS (32 data + pad), 80B = 5*16B
#define H1_PITCH 136   // halves per h1 row in LDS (128 data + pad), 272B = 17*16B

__global__ __launch_bounds__(256, 2)
void topkast_mlp_fused_kernel(const float* __restrict__ X,
                              const _Float16* __restrict__ W1B,
                              const float* __restrict__ b1,
                              const _Float16* __restrict__ W2B,
                              const float* __restrict__ b2,
                              const float* __restrict__ W3,
                              const float* __restrict__ b3,
                              float* __restrict__ out) {
    __shared__ __align__(16) _Float16 Xs[128 * XS_PITCH];
    __shared__ __align__(16) _Float16 H1[128 * H1_PITCH];

    const int row0 = blockIdx.x * 128;
    const int tid  = threadIdx.x;

    // ---- stage X tile -> LDS as f16, K-padded to 32 (cols 13..39 = 0) ----
    for (int i = tid; i < 128 * XS_PITCH; i += 256) {
        int r = i / XS_PITCH;
        int c = i - r * XS_PITCH;
        float v = (c < 13) ? X[(size_t)(row0 + r) * 13 + c] : 0.f;
        Xs[i] = (_Float16)v;
    }
    __syncthreads();

    const int lane    = tid & 31;
    const int wave    = tid >> 5;
    const int m       = lane & 15;          // A-layout row / C-layout column
    const int halfsel = (lane >= 16) ? 1 : 0;
    const int kb      = halfsel * 16;       // A-layout K base: lane<16 -> 0, else 16
    const int lrow    = wave * 16;          // this wave's local row base

    // ---- layer 1: h1 = relu(X @ W1m^T + b1), 8 WMMAs, inline-0 C ----
    {
        const _Float16* arow = &Xs[(lrow + m) * XS_PITCH];
        h8 c0 = *(const h8*)(arow + kb);        // K = kb..kb+7
        h8 c1 = *(const h8*)(arow + kb + 8);    // K = kb+8..kb+15
        v16h a1 = __builtin_shufflevector(c0, c1, 0, 1, 2, 3, 4, 5, 6, 7,
                                                  8, 9, 10, 11, 12, 13, 14, 15);
        #pragma unroll
        for (int t = 0; t < 8; ++t) {
            v16h bf = *(const v16h*)(W1B + ((t * 32 + lane) << 4));
            v8f acc = {};                       // inline SRC2 = 0
            acc = WMMA_F16(a1, bf, acc);
            float bv = b1[t * 16 + m];
            #pragma unroll
            for (int r = 0; r < 8; ++r) {
                float v = relu_med3(acc[r] + bv);
                H1[(lrow + r + halfsel * 8) * H1_PITCH + t * 16 + m] = (_Float16)v;
            }
        }
    }
    __syncthreads();

    // ---- layer 2 + 3: h2 = relu(h1 @ W2m^T + b2); out = h2 @ W3^T + b3 ----
    // A-fragments of h1 are invariant across the 8 n-tiles: load once.
    v16h afrag[4];
    {
        const _Float16* h1row = &H1[(lrow + m) * H1_PITCH];
        #pragma unroll
        for (int kc = 0; kc < 4; ++kc) {
            int kb2 = kc * 32 + kb;
            h8 x0 = *(const h8*)(h1row + kb2);
            h8 x1 = *(const h8*)(h1row + kb2 + 8);
            afrag[kc] = __builtin_shufflevector(x0, x1, 0, 1, 2, 3, 4, 5, 6, 7,
                                                        8, 9, 10, 11, 12, 13, 14, 15);
        }
    }

    float part[8] = {0.f, 0.f, 0.f, 0.f, 0.f, 0.f, 0.f, 0.f};
    #pragma unroll
    for (int t = 0; t < 8; ++t) {
        // batch the 4 B-fragment loads so they clause and overlap the epilogue
        v16h bf[4];
        #pragma unroll
        for (int kc = 0; kc < 4; ++kc)
            bf[kc] = *(const v16h*)(W2B + (((t * 4 + kc) * 32 + lane) << 4));
        v8f acc = {};                           // inline SRC2 = 0
        #pragma unroll
        for (int kc = 0; kc < 4; ++kc)
            acc = WMMA_F16(afrag[kc], bf[kc], acc);
        float bv  = b2[t * 16 + m];
        float w3v = W3[t * 16 + m];
        #pragma unroll
        for (int r = 0; r < 8; ++r)
            part[r] = fmaf(relu_med3(acc[r] + bv), w3v, part[r]);   // relu + layer-3 dot
    }

    // reduce across the 16 lanes of each half (N dimension of h2)
    #pragma unroll
    for (int mask = 1; mask < 16; mask <<= 1) {
        #pragma unroll
        for (int r = 0; r < 8; ++r)
            part[r] += __shfl_xor(part[r], mask, 32);
    }
    if (m == 0) {   // lane 0 -> rows lrow..lrow+7, lane 16 -> rows lrow+8..lrow+15
        float bb = b3[0];
        int gr = row0 + lrow + halfsel * 8;
        #pragma unroll
        for (int r = 0; r < 8; ++r)
            out[gr + r] = part[r] + bb;
    }
}

// ---------------------------------------------------------------------------
// Launch
// ---------------------------------------------------------------------------
extern "C" void kernel_launch(void* const* d_in, const int* in_sizes, int n_in,
                              void* d_out, int out_size, void* d_ws, size_t ws_size,
                              hipStream_t stream) {
    const float* X  = (const float*)d_in[0];
    const float* W1 = (const float*)d_in[1];
    const float* b1 = (const float*)d_in[2];
    const float* W2 = (const float*)d_in[3];
    const float* b2 = (const float*)d_in[4];
    const float* W3 = (const float*)d_in[5];
    const float* b3 = (const float*)d_in[6];
    float* out = (float*)d_out;

    const int N = in_sizes[0] / 13;   // rows

    // workspace layout (32B-aligned fragment bases for clean v16h loads)
    unsigned* thr  = (unsigned*)d_ws;                          // 2 x u32
    _Float16* W1B  = (_Float16*)((char*)d_ws + 64);            // 4096 halves
    _Float16* W2B  = (_Float16*)((char*)d_ws + 64 + 8192);     // 16384 halves
    (void)ws_size; (void)n_in; (void)out_size;

    // k = max(1, ceil((1 - p/100) * numel)):  W1: ceil(0.8*1664)=1332; W2: 8192
    topkast_threshold_kernel<<<1, 256, 0, stream>>>(W1, 128 * 13, 1332, thr + 0);
    topkast_threshold_kernel<<<1, 256, 0, stream>>>(W2, 128 * 128, 8192, thr + 1);

    topkast_pack_kernel<<<80, 256, 0, stream>>>(W1, W2, thr, W1B, W2B);

    const int nblocks = (N + 127) / 128;
    topkast_mlp_fused_kernel<<<nblocks, 256, 0, stream>>>(X, W1B, b1, W2B, b2, W3, b3, out);
}
